// AMS_18975165514201
// MI455X (gfx1250) — matrix operand order repaired and checked by reading, not compiled, not run on hardware
//
#include <hip/hip_runtime.h>
#include <hip/hip_bf16.h>

typedef __attribute__((ext_vector_type(16))) _Float16 v16h;
typedef __attribute__((ext_vector_type(8)))  float    v8f;
typedef __attribute__((ext_vector_type(4)))  float    v4f;

#define BB 64
#define LL 336
#define NN 64
#define DD 32
#define EE 4
#define NFREQ 167   // valid frequency indices 1..167 (t even -> drop DC and Nyquist)
#define MT 11       // 176 rows of DFT matrix (16-row tiles)
#define KT 11       // 352 padded K (32-col tiles)
#define TPAD 352
#define PI_F 3.14159265358979f

// ---------------------------------------------------------------------------
// Kernel 1: extract x3, moving-avg trend w/ softmax mix, write base = x3+trend,
// and write x3 as f16 pre-swizzled into WMMA B-matrix (K x N) lane layout.
// The gather sweeps all of x (one float per 128B line) -> x becomes L2-resident
// (176 MB vs 192 MB L2) for reuse by the final scale kernel.
// ---------------------------------------------------------------------------
__global__ void AMS_decomp_kernel(const float* __restrict__ x,
                                  const float* __restrict__ decomp_w,
                                  const float* __restrict__ decomp_b,
                                  float* __restrict__ base,
                                  _Float16* __restrict__ x3t) {
    __shared__ float tile[LL * 16];
    int b  = blockIdx.x >> 2;
    int ng = blockIdx.x & 3;        // n-tile (16 columns)
    int n0 = ng * 16;

    for (int i = threadIdx.x; i < LL * 16; i += blockDim.x) {
        int t = i >> 4, nl = i & 15;
        tile[i] = x[(((size_t)b * LL + t) * NN + (n0 + nl)) * DD];
    }
    __syncthreads();

    float w0 = decomp_w[0], w1 = decomp_w[1], w2 = decomp_w[2];
    float c0 = decomp_b[0], c1 = decomp_b[1], c2 = decomp_b[2];

    for (int i = threadIdx.x; i < TPAD * 16; i += blockDim.x) {
        int t = i >> 4, nl = i & 15;
        int kiter = t >> 5, klocal = t & 31;
        // f16 B-matrix (32x16) layout: lanes 0-15 hold K {0..7,16..23}, lanes 16-31 hold K {8..15,24..31}
        int lane = nl + (((klocal >> 3) & 1) << 4);
        int j    = (klocal & 7) + ((klocal >> 4) << 3);
        size_t tidx = ((((size_t)b * 4 + ng) * KT + kiter) * 32 + lane) * 16 + j;
        if (t >= LL) { x3t[tidx] = (_Float16)0.f; continue; }

        float v = tile[(t << 4) + nl];

        float ma[3];
        const int ks[3] = {4, 8, 12};
        #pragma unroll
        for (int q = 0; q < 3; ++q) {
            int k = ks[q];
            int front = k - 1 - (k - 1) / 2;   // edge-replicate padding
            float s = 0.f;
            for (int jj = 0; jj < k; ++jj) {
                int tt = t + jj - front;
                tt = tt < 0 ? 0 : (tt > LL - 1 ? LL - 1 : tt);
                s += tile[(tt << 4) + nl];
            }
            ma[q] = s / (float)k;
        }
        float l0 = v * w0 + c0, l1 = v * w1 + c1, l2 = v * w2 + c2;
        float mx = fmaxf(l0, fmaxf(l1, l2));
        float e0 = expf(l0 - mx), e1 = expf(l1 - mx), e2 = expf(l2 - mx);
        float inv = 1.f / (e0 + e1 + e2);
        float trend = (ma[0] * e0 + ma[1] * e1 + ma[2] * e2) * inv;

        base[((size_t)b * LL + t) * NN + (n0 + nl)] = v + trend;
        x3t[tidx] = (_Float16)v;
    }
}

// ---------------------------------------------------------------------------
// Kernel 2: DFT coefficient matrices in f16, pre-swizzled to the 16-bit
// A-matrix (16x32) per-lane layout. Row m <-> frequency m+1. Ac = cos,
// As = -sin (so Xi = Sum x * As).
// ---------------------------------------------------------------------------
__global__ void AMS_dftcoef_kernel(_Float16* __restrict__ Ac,
                                   _Float16* __restrict__ As) {
    int id = blockIdx.x * blockDim.x + threadIdx.x;
    if (id >= MT * KT * 512) return;
    int j     = id & 15;
    int lane  = (id >> 4) & 31;
    int kiter = (id >> 9) % KT;
    int mtile = id / (KT * 512);
    int m = mtile * 16 + (lane & 15);
    // A layout: lane<16 -> K {0..7,16..23}; lane>=16 -> K {8..15,24..31}
    int k = kiter * 32 + j + (j < 8 ? 0 : 8) + (lane < 16 ? 0 : 8);
    float c = 0.f, s = 0.f;
    if (m < NFREQ && k < LL) {
        float th = 2.f * PI_F * (float)(m + 1) * (float)k / (float)LL;
        c = cosf(th);
        s = -sinf(th);
    }
    Ac[id] = (_Float16)c;
    As[id] = (_Float16)s;
}

// ---------------------------------------------------------------------------
// Kernel 3: batched DFT-as-GEMM with WMMA. One block per (b, mtile),
// 4 full waves (wave = n-tile). Two v_wmma_f32_16x16x32_f16 per K step.
// ---------------------------------------------------------------------------
__global__ void AMS_dft_wmma_kernel(const _Float16* __restrict__ x3t,
                                    const _Float16* __restrict__ Ac,
                                    const _Float16* __restrict__ As,
                                    float* __restrict__ Xr,
                                    float* __restrict__ Xi) {
    int b     = blockIdx.x / MT;
    int mtile = blockIdx.x % MT;
    int wave  = threadIdx.x >> 5;   // n-tile
    int lane  = threadIdx.x & 31;

    const v16h* Bp = (const v16h*)x3t;
    const v16h* Cp = (const v16h*)Ac;
    const v16h* Sp = (const v16h*)As;

    v8f accr = {};
    v8f acci = {};
    for (int kiter = 0; kiter < KT; ++kiter) {
        v16h a_c = Cp[((size_t)mtile * KT + kiter) * 32 + lane];
        v16h a_s = Sp[((size_t)mtile * KT + kiter) * 32 + lane];
        v16h bm  = Bp[(((size_t)b * 4 + wave) * KT + kiter) * 32 + lane];
        accr = __builtin_amdgcn_wmma_f32_16x16x32_f16(false, a_c, false, bm,
                                                      (short)0, accr, false, false);
        acci = __builtin_amdgcn_wmma_f32_16x16x32_f16(false, a_s, false, bm,
                                                      (short)0, acci, false, false);
    }
    // C/D layout: VGPR r -> M = r (+8 for lanes 16-31), N = lane&15
    int n     = wave * 16 + (lane & 15);
    int mbase = mtile * 16 + ((lane >> 4) << 3);
    #pragma unroll
    for (int r = 0; r < 8; ++r) {
        int m = mbase + r;
        if (m < NFREQ) {
            Xr[((size_t)b * NFREQ + m) * NN + n] = accr[r];
            Xi[((size_t)b * NFREQ + m) * NN + n] = acci[r];
        }
    }
}

// ---------------------------------------------------------------------------
// Kernel 4: per (b,n) top-3 |X(f)| and resynthesis parameters
// seasonality(t) = Sum_j coef_j * cos(w_j * t + phase_j), coef = 2|X|/L
// ---------------------------------------------------------------------------
__global__ void AMS_topk_kernel(const float* __restrict__ Xr,
                                const float* __restrict__ Xi,
                                float* __restrict__ params) {
    int id = blockIdx.x * blockDim.x + threadIdx.x;
    if (id >= BB * NN) return;
    int b = id / NN, n = id % NN;
    float v0 = -1.f, v1 = -1.f, v2 = -1.f;
    int   i0 = 0, i1 = 0, i2 = 0;
    for (int m = 0; m < NFREQ; ++m) {
        float xr = Xr[((size_t)b * NFREQ + m) * NN + n];
        float xi = Xi[((size_t)b * NFREQ + m) * NN + n];
        float a = xr * xr + xi * xi;
        if (a > v0)      { v2 = v1; i2 = i1; v1 = v0; i1 = i0; v0 = a; i0 = m; }
        else if (a > v1) { v2 = v1; i2 = i1; v1 = a; i1 = m; }
        else if (a > v2) { v2 = a; i2 = m; }
    }
    int idxs[3] = {i0, i1, i2};
    #pragma unroll
    for (int j = 0; j < 3; ++j) {
        int m = idxs[j];
        float xr = Xr[((size_t)b * NFREQ + m) * NN + n];
        float xi = Xi[((size_t)b * NFREQ + m) * NN + n];
        float amp = sqrtf(xr * xr + xi * xi);
        params[(size_t)id * 9 + j * 3 + 0] = 2.f * amp / (float)LL; // conj pair doubles
        params[(size_t)id * 9 + j * 3 + 1] = 2.f * PI_F * (float)(m + 1) / (float)LL;
        params[(size_t)id * 9 + j * 3 + 2] = atan2f(xi, xr);
    }
}

// ---------------------------------------------------------------------------
// Kernel 5: g[b,t] = Sum_n (base + seasonality) * start_w[n] + start_b
// ---------------------------------------------------------------------------
__global__ void AMS_g_kernel(const float* __restrict__ base,
                             const float* __restrict__ params,
                             const float* __restrict__ start_w,
                             const float* __restrict__ start_b,
                             float* __restrict__ g) {
    int b = blockIdx.x;
    int t = threadIdx.x;
    if (t >= LL) return;
    float acc = 0.f;
    float tf = (float)t;
    for (int n = 0; n < NN; ++n) {
        const float* p = params + ((size_t)b * NN + n) * 9;
        float seas = 0.f;
        #pragma unroll
        for (int j = 0; j < 3; ++j)
            seas += p[j * 3] * cosf(p[j * 3 + 1] * tf + p[j * 3 + 2]);
        acc += (base[((size_t)b * LL + t) * NN + n] + seas) * start_w[n];
    }
    g[b * LL + t] = acc + start_b[0];
}

// ---------------------------------------------------------------------------
// Kernel 6: logits. One thread per (b,e): dot(g[b,:], w[:,e]).
// ---------------------------------------------------------------------------
__global__ void AMS_logits_kernel(const float* __restrict__ g,
                                  const float* __restrict__ wg_w,
                                  const float* __restrict__ wg_b,
                                  const float* __restrict__ wn_w,
                                  const float* __restrict__ wn_b,
                                  const float* __restrict__ noise,
                                  float* __restrict__ clean,
                                  float* __restrict__ nstd,
                                  float* __restrict__ noisy) {
    int id = threadIdx.x;          // 0..255 = (b,e)
    int b = id >> 2, e = id & 3;
    float sc = 0.f, sn = 0.f;
    for (int t = 0; t < LL; ++t) {
        float gv = g[b * LL + t];
        sc += gv * wg_w[t * EE + e];
        sn += gv * wn_w[t * EE + e];
    }
    float cl = sc + wg_b[e];
    float z  = sn + wn_b[e];
    float sp = fmaxf(z, 0.f) + log1pf(expf(-fabsf(z)));  // stable softplus
    float sd = sp + 0.01f;
    clean[id] = cl;
    nstd[id]  = sd;
    noisy[id] = cl + noise[id] * sd;
}

// ---------------------------------------------------------------------------
// Kernel 7: noisy-top-k gates, load/importance, cv^2 balance loss.
// Single block, 64 threads (one per batch row), LDS reduction.
// ---------------------------------------------------------------------------
__global__ void AMS_gates_kernel(const float* __restrict__ clean,
                                 const float* __restrict__ nstd,
                                 const float* __restrict__ noisy,
                                 float* __restrict__ gate_sum,
                                 float* __restrict__ loss_out) {
    __shared__ float red[8];   // [0..3] importance, [4..7] load
    int b = threadIdx.x;
    if (b < 8) red[b] = 0.f;
    __syncthreads();
    if (b < BB) {
        float v[4]; int idx[4];
        #pragma unroll
        for (int e = 0; e < 4; ++e) { v[e] = noisy[b * 4 + e]; idx[e] = e; }
        #define AMS_SW(i, j) if (v[i] < v[j]) { float tv = v[i]; v[i] = v[j]; v[j] = tv; \
                                                int ti = idx[i]; idx[i] = idx[j]; idx[j] = ti; }
        AMS_SW(0, 1) AMS_SW(2, 3) AMS_SW(0, 2) AMS_SW(1, 3) AMS_SW(1, 2)
        #undef AMS_SW
        float e1 = expf(v[1] - v[0]);
        float g0 = 1.f / (1.f + e1);
        float g1 = e1 / (1.f + e1);
        gate_sum[b] = g0 + g1;
        atomicAdd(&red[idx[0]], g0);
        atomicAdd(&red[idx[1]], g1);
        float thr_in = v[2], thr_out = v[1];   // K=2
        #pragma unroll
        for (int e = 0; e < 4; ++e) {
            float cl = clean[b * 4 + e], sd = nstd[b * 4 + e], nz = noisy[b * 4 + e];
            float arg = (nz > thr_in) ? (cl - thr_in) / sd : (cl - thr_out) / sd;
            float prob = 0.5f * erfcf(-arg * 0.70710678118f);  // normal cdf
            atomicAdd(&red[4 + e], prob);
        }
    }
    __syncthreads();
    if (b == 0) {
        float cv[2];
        #pragma unroll
        for (int q = 0; q < 2; ++q) {
            float s = 0.f;
            for (int e = 0; e < 4; ++e) s += red[q * 4 + e];
            float mean = s * 0.25f;
            float var = 0.f;
            for (int e = 0; e < 4; ++e) { float d = red[q * 4 + e] - mean; var += d * d; }
            var /= 3.f;   // ddof=1
            cv[q] = var / (mean * mean + 1e-10f);
        }
        *loss_out = 0.01f * (cv[0] + cv[1]);
    }
}

// ---------------------------------------------------------------------------
// Kernel 8: output = x * gate_sum[b]; bandwidth-bound 16B-vector sweep.
// x reads should mostly hit L2 (x made resident by kernel 1's sweep).
// Output stores are non-temporal (clang ext-vector type, legal for the
// builtin) so the 176 MB write stream does not evict the still-unread
// portions of x from L2. Prefetch next chunk of x (global_prefetch_b8).
// ---------------------------------------------------------------------------
__global__ void AMS_scale_kernel(const v4f* __restrict__ x,
                                 const float* __restrict__ gate_sum,
                                 v4f* __restrict__ out) {
    int b = blockIdx.y;
    size_t i = (size_t)b * (LL * NN * DD / 4) +
               (size_t)blockIdx.x * blockDim.x + threadIdx.x;
    __builtin_prefetch(&x[i + 256], 0, 3);   // next block's chunk
    float s = gate_sum[b];
    v4f v = x[i] * s;
    __builtin_nontemporal_store(v, &out[i]);
}

// ---------------------------------------------------------------------------
extern "C" void kernel_launch(void* const* d_in, const int* in_sizes, int n_in,
                              void* d_out, int out_size, void* d_ws, size_t ws_size,
                              hipStream_t stream) {
    const float* x        = (const float*)d_in[0];
    const float* noise    = (const float*)d_in[2];
    const float* start_w  = (const float*)d_in[3];
    const float* start_b  = (const float*)d_in[4];
    const float* wg_w     = (const float*)d_in[5];
    const float* wg_b     = (const float*)d_in[6];
    const float* wn_w     = (const float*)d_in[7];
    const float* wn_b     = (const float*)d_in[8];
    const float* decomp_w = (const float*)d_in[9];
    const float* decomp_b = (const float*)d_in[10];

    char* p = (char*)d_ws;
    auto alloc = [&](size_t bytes) -> void* {
        void* r = (void*)p;
        p += (bytes + 255) & ~(size_t)255;
        return r;
    };
    float*     base     = (float*)alloc((size_t)BB * LL * NN * 4);
    _Float16*  x3t      = (_Float16*)alloc((size_t)BB * 4 * KT * 512 * 2);
    _Float16*  Ac       = (_Float16*)alloc((size_t)MT * KT * 512 * 2);
    _Float16*  As       = (_Float16*)alloc((size_t)MT * KT * 512 * 2);
    float*     Xr       = (float*)alloc((size_t)BB * NFREQ * NN * 4);
    float*     Xi       = (float*)alloc((size_t)BB * NFREQ * NN * 4);
    float*     params   = (float*)alloc((size_t)BB * NN * 9 * 4);
    float*     g        = (float*)alloc((size_t)BB * LL * 4);
    float*     clean    = (float*)alloc(256 * 4);
    float*     nstd     = (float*)alloc(256 * 4);
    float*     noisy    = (float*)alloc(256 * 4);
    float*     gate_sum = (float*)alloc(BB * 4);

    float* out      = (float*)d_out;
    float* loss_out = out + (size_t)BB * LL * NN * DD;  // out_size - 1

    AMS_decomp_kernel<<<dim3(BB * 4), 256, 0, stream>>>(x, decomp_w, decomp_b, base, x3t);
    AMS_dftcoef_kernel<<<dim3((MT * KT * 512 + 255) / 256), 256, 0, stream>>>(Ac, As);
    AMS_dft_wmma_kernel<<<dim3(BB * MT), 128, 0, stream>>>(x3t, Ac, As, Xr, Xi);
    AMS_topk_kernel<<<dim3((BB * NN + 255) / 256), 256, 0, stream>>>(Xr, Xi, params);
    AMS_g_kernel<<<dim3(BB), 352, 0, stream>>>(base, params, start_w, start_b, g);
    AMS_logits_kernel<<<dim3(1), 256, 0, stream>>>(g, wg_w, wg_b, wn_w, wn_b, noise,
                                                   clean, nstd, noisy);
    AMS_gates_kernel<<<dim3(1), 64, 0, stream>>>(clean, nstd, noisy, gate_sum, loss_out);
    AMS_scale_kernel<<<dim3(LL * NN * DD / 4 / 256, BB), 256, 0, stream>>>(
        (const v4f*)x, gate_sum, (v4f*)d_out);
}
